// TransAttention_317827580568
// MI455X (gfx1250) — matrix operand order repaired and checked by reading, not compiled, not run on hardware
//
#include <hip/hip_runtime.h>
#include <hip/hip_bf16.h>

typedef __attribute__((ext_vector_type(16))) _Float16 v16h;
typedef __attribute__((ext_vector_type(8)))  _Float16 v8h;
typedef __attribute__((ext_vector_type(8)))  float    v8f;

#define BM 128
#define BN 128
#define BK 64
#define LDT 72   // padded LDS row stride in halfs (144 B = 9*16B, keeps 16B chunks aligned)

__device__ __forceinline__ v16h make_v16(v8h lo, v8h hi) {
    return __builtin_shufflevector(lo, hi, 0,1,2,3,4,5,6,7,8,9,10,11,12,13,14,15);
}

// CDNA5 async global -> LDS copy of 16 bytes per lane (ASYNCcnt-tracked).
__device__ __forceinline__ void async_cp16(unsigned int lds_off, const _Float16* gptr) {
    asm volatile("global_load_async_to_lds_b128 %0, %1, off"
                 :: "v"(lds_off), "v"(gptr) : "memory");
}
__device__ __forceinline__ void wait_async_le8() {
    asm volatile("s_wait_asynccnt 0x8" ::: "memory");
}
__device__ __forceinline__ void wait_async_0() {
    asm volatile("s_wait_asynccnt 0x0" ::: "memory");
}

// Batched WMMA GEMM: C[b, r, n] = act( alpha * sum_k A[b,r,k] * B[b][n*ldb + k] + bias[n] )
// (B is always "m-major": contraction index contiguous — torch Linear weight layout.)
__global__ __launch_bounds__(256)
void wmma_gemm(const _Float16* __restrict__ A, long strideA,
               const _Float16* __restrict__ Bp, long strideB,
               _Float16* __restrict__ C, long strideC,
               const float* __restrict__ bias,
               int N, int K, int lda, int ldb,
               float alpha, int selu)
{
    // layout (halfs): [ A buf0 | B buf0 | A buf1 | B buf1 ], each BM*LDT
    __shared__ __align__(16) _Float16 smem[4 * BM * LDT];
    const unsigned int ldsBase = (unsigned int)(uintptr_t)(void*)smem;
    const unsigned int BUFB = (unsigned int)(2 * BM * LDT * 2);   // bytes per buffer pair
    const unsigned int BOFF = (unsigned int)(BM * LDT * 2);       // bytes from A to B

    const int bz = blockIdx.z;
    const int n0 = blockIdx.x * BN;
    const _Float16* Ag = A  + (long)bz * strideA + (long)(blockIdx.y * BM) * lda;
    const _Float16* Bg = Bp + (long)bz * strideB + (long)n0 * ldb;
    _Float16*       Cb = C  + (long)bz * strideC;

    const int tid  = threadIdx.x;
    const int wave = tid >> 5;
    const int lane = tid & 31;
    const int wm   = wave >> 2;   // 0..1 : 64-row slab
    const int wn   = wave & 3;    // 0..3 : 32-col slab
    const int lm   = lane & 15;
    const int lk   = lane >> 4;

    v8f acc[4][2];
    const v8f vzero = {0.f,0.f,0.f,0.f,0.f,0.f,0.f,0.f};
#pragma unroll
    for (int mi = 0; mi < 4; ++mi)
#pragma unroll
        for (int ni = 0; ni < 2; ++ni) acc[mi][ni] = vzero;

    const int nstage = K / BK;

    // prologue: stage tile 0 into buffer 0
#pragma unroll
    for (int it = 0; it < 4; ++it) {
        int idx = tid + it * 256;                 // 1024 chunks of 8 halfs (128 x 64)
        int r = idx >> 3;
        int c = (idx & 7) * 8;
        unsigned int off = (unsigned int)(r * LDT + c) * 2u;
        async_cp16(ldsBase + off,        Ag + (long)r * lda + c);
        async_cp16(ldsBase + BOFF + off, Bg + (long)r * ldb + c);
    }

    for (int s = 0; s < nstage; ++s) {
        const int cur = s & 1;
        if (s + 1 < nstage) {
            const unsigned int nb = (cur ^ 1) ? BUFB : 0u;
            const long k1 = (long)(s + 1) * BK;
#pragma unroll
            for (int it = 0; it < 4; ++it) {
                int idx = tid + it * 256;
                int r = idx >> 3;
                int c = (idx & 7) * 8;
                unsigned int off = (unsigned int)(r * LDT + c) * 2u;
                async_cp16(ldsBase + nb + off,        Ag + (long)r * lda + k1 + c);
                async_cp16(ldsBase + nb + BOFF + off, Bg + (long)r * ldb + k1 + c);
            }
            wait_async_le8();   // current tile's 8 async ops have landed
        } else {
            wait_async_0();
        }
        __syncthreads();

        const _Float16* As = smem + (cur ? 2 * BM * LDT : 0);
        const _Float16* Bs = As + BM * LDT;
#pragma unroll
        for (int j = 0; j < 2; ++j) {           // two 16x16x32 K-steps per stage
            v16h af[4], bf[2];
#pragma unroll
            for (int mi = 0; mi < 4; ++mi) {
                const _Float16* p = &As[(wm * 64 + mi * 16 + lm) * LDT + j * 32 + lk * 8];
                af[mi] = make_v16(*(const v8h*)(p), *(const v8h*)(p + 16));
            }
#pragma unroll
            for (int ni = 0; ni < 2; ++ni) {
                const _Float16* p = &Bs[(wn * 32 + ni * 16 + lm) * LDT + j * 32 + lk * 8];
                bf[ni] = make_v16(*(const v8h*)(p), *(const v8h*)(p + 16));
            }
#pragma unroll
            for (int mi = 0; mi < 4; ++mi)
#pragma unroll
                for (int ni = 0; ni < 2; ++ni)
                    acc[mi][ni] = __builtin_amdgcn_wmma_f32_16x16x32_f16(
                        false, af[mi], false, bf[ni], (short)0, acc[mi][ni], false, false);
        }
        __syncthreads();
    }

    // epilogue: D layout — VGPR r: row = lk*8 + r, col = lm
    const float kSeluScale = 1.0507009873554805f;
    const float kSeluAlpha = 1.6732632423543772f;
#pragma unroll
    for (int mi = 0; mi < 4; ++mi) {
        int row = blockIdx.y * BM + wm * 64 + mi * 16 + lk * 8;
#pragma unroll
        for (int ni = 0; ni < 2; ++ni) {
            int col = n0 + wn * 32 + ni * 16 + lm;
            float bv = bias ? bias[col] : 0.0f;
#pragma unroll
            for (int r = 0; r < 8; ++r) {
                float v = acc[mi][ni][r] * alpha + bv;
                if (selu) v = (v > 0.0f) ? kSeluScale * v
                                         : kSeluScale * kSeluAlpha * (__expf(v) - 1.0f);
                Cb[(long)(row + r) * N + col] = (_Float16)v;
            }
        }
    }
}

// [B,Nn,Cc] f32 -> [B,Cc,Nn] f16
__global__ void transpose_cvt(const float* __restrict__ in, _Float16* __restrict__ out,
                              int Nn, int Cc)
{
    __shared__ float tile[32][33];
    int b = blockIdx.z, n0 = blockIdx.y * 32, c0 = blockIdx.x * 32;
    int tx = threadIdx.x, ty = threadIdx.y;   // 32 x 8
    const float* ip = in + (long)b * Nn * Cc;
#pragma unroll
    for (int i = 0; i < 32; i += 8)
        tile[ty + i][tx] = ip[(long)(n0 + ty + i) * Cc + c0 + tx];
    __syncthreads();
    _Float16* op = out + (long)b * Nn * Cc;
#pragma unroll
    for (int i = 0; i < 32; i += 8)
        op[(long)(c0 + ty + i) * Nn + n0 + tx] = (_Float16)tile[tx][ty + i];
}

__global__ void cvt_f32_f16(const float* __restrict__ in, _Float16* __restrict__ out, long n)
{
    long i = (long)blockIdx.x * blockDim.x + threadIdx.x;
    long stride = (long)gridDim.x * blockDim.x;
    for (; i < n; i += stride) out[i] = (_Float16)in[i];
}

// softmax over the last dim, one block per row of length L
__global__ void softmax_rows(const _Float16* __restrict__ x, _Float16* __restrict__ y, int L)
{
    long row = blockIdx.x;
    const _Float16* xr = x + row * L;
    _Float16*       yr = y + row * L;
    int t = threadIdx.x;
    __shared__ float red[256];
    float m = -1e30f;
    for (int i = t; i < L; i += 256) m = fmaxf(m, (float)xr[i]);
    red[t] = m; __syncthreads();
    for (int s = 128; s > 0; s >>= 1) { if (t < s) red[t] = fmaxf(red[t], red[t + s]); __syncthreads(); }
    m = red[0]; __syncthreads();
    float sum = 0.0f;
    for (int i = t; i < L; i += 256) sum += __expf((float)xr[i] - m);
    red[t] = sum; __syncthreads();
    for (int s = 128; s > 0; s >>= 1) { if (t < s) red[t] += red[t + s]; __syncthreads(); }
    float inv = 1.0f / red[0];
    for (int i = t; i < L; i += 256) yr[i] = (_Float16)(__expf((float)xr[i] - m) * inv);
}

// softmax over the channel dim of [B, Cc, Nn]; one thread per (b, n)
__global__ void softmax_cols(const _Float16* __restrict__ x, _Float16* __restrict__ y,
                             int Cc, int Nn)
{
    int b = blockIdx.y;
    int n = blockIdx.x * blockDim.x + threadIdx.x;
    const _Float16* xb = x + (long)b * Cc * Nn;
    _Float16*       yb = y + (long)b * Cc * Nn;
    float m = -1e30f;
    for (int c = 0; c < Cc; ++c) m = fmaxf(m, (float)xb[(long)c * Nn + n]);
    float s = 0.0f;
    for (int c = 0; c < Cc; ++c) s += __expf((float)xb[(long)c * Nn + n] - m);
    float inv = 1.0f / s;
    for (int c = 0; c < Cc; ++c)
        yb[(long)c * Nn + n] = (_Float16)(__expf((float)xb[(long)c * Nn + n] - m) * inv);
}

// y: [B,Cc,Nn] f16 ; src/out: [B,Nn,Cc] f32 ; out = l2norm_over_C(y^T)*0.1 + src*blend
__global__ void finalize_k(const _Float16* __restrict__ y, const float* __restrict__ src,
                           float* __restrict__ out, int Cc, int Nn, float blend)
{
    int b = blockIdx.y, n = blockIdx.x, t = threadIdx.x;
    const _Float16* yb = y + (long)b * Cc * Nn + n;
    __shared__ float red[256];
    float s = 0.0f;
    for (int c = t; c < Cc; c += 256) { float v = (float)yb[(long)c * Nn]; s += v * v; }
    red[t] = s; __syncthreads();
    for (int st = 128; st > 0; st >>= 1) { if (t < st) red[t] += red[t + st]; __syncthreads(); }
    float inv = 0.1f / fmaxf(sqrtf(red[0]), 1e-12f);
    const float* sp = src + ((long)b * Nn + n) * Cc;
    float*       op = out + ((long)b * Nn + n) * Cc;
    for (int c = t; c < Cc; c += 256)
        op[c] = (float)yb[(long)c * Nn] * inv + sp[c] * blend;
}

// ---------------- host side ----------------

static void launch_gemm(const _Float16* A, long sA, const _Float16* B, long sB,
                        _Float16* C, long sC, const float* bias,
                        int R, int N, int K, int lda, int ldb,
                        float alpha, int selu, int batch, hipStream_t stream)
{
    dim3 grid(N / BN, R / BM, batch);
    wmma_gemm<<<grid, 256, 0, stream>>>(A, sA, B, sB, C, sC, bias,
                                        N, K, lda, ldb, alpha, selu);
}

static void run_fc1(const _Float16* in, _Float16* tA, _Float16* tB, _Float16* outp,
                    const _Float16* Wh, const float* bias, int R, hipStream_t stream)
{
    const _Float16* cur = in;
    for (int i = 0; i < 7; ++i) {
        _Float16* dst = (i == 6) ? outp : ((i & 1) ? tB : tA);
        launch_gemm(cur, 0, Wh + (long)i * 1024 * 1024, 0, dst, 0,
                    bias + (long)i * 1024, R, 1024, 1024, 1024, 1024,
                    1.0f, (i == 3) ? 1 : 0, 1, stream);
        cur = dst;
    }
}

extern "C" void kernel_launch(void* const* d_in, const int* in_sizes, int n_in,
                              void* d_out, int out_size, void* d_ws, size_t ws_size,
                              hipStream_t stream)
{
    (void)in_sizes; (void)n_in; (void)out_size; (void)ws_size;
    const long NH = 1024, CC = 512, BB = 16;
    const long SZX = BB * CC * NH;          // 8,388,608 halfs per activation tensor
    const long SZW = 7 * NH * NH;           // 7,340,032 halfs

    const float* pseudo = (const float*)d_in[0];   // [16,1024,512]
    const float* valid  = (const float*)d_in[1];   // [16,1024,512]
    const float* W      = (const float*)d_in[2];   // [7,1024,1024]
    const float* bias   = (const float*)d_in[3];   // [7,1024]
    float* out = (float*)d_out;                    // p_end then v_end, each [16,1024,512]

    _Float16* p = (_Float16*)d_ws;
    _Float16* Wh  = p; p += SZW;
    _Float16* xp  = p; p += SZX;   // [B,C,N] transposed pseudo
    _Float16* xv  = p; p += SZX;   // [B,C,N] transposed valid
    _Float16* xpT = p; p += SZX;   // [B,N,C] raw pseudo (f16) — m-major B for att@pse
    _Float16* xvT = p; p += SZX;   // [B,N,C] raw valid (f16)
    _Float16* t0  = p; p += SZX;
    _Float16* t1  = p; p += SZX;
    _Float16* pqk = p; p += SZX;   // later reused as the attention matrix
    _Float16* vqk = p; p += SZX;   // later reused as branch activation buffer
    _Float16* sQp = p; p += SZX;
    _Float16* sKp = p; p += SZX;
    _Float16* sQv = p; p += SZX;
    _Float16* sKv = p; p += SZX;

    // 1) convert weights + inputs (both layouts)
    cvt_f32_f16<<<4096, 256, 0, stream>>>(W, Wh, SZW);
    cvt_f32_f16<<<4096, 256, 0, stream>>>(pseudo, xpT, SZX);
    cvt_f32_f16<<<4096, 256, 0, stream>>>(valid,  xvT, SZX);
    dim3 tg((unsigned)(CC / 32), (unsigned)(NH / 32), (unsigned)BB);
    transpose_cvt<<<tg, dim3(32, 8), 0, stream>>>(pseudo, xp, (int)NH, (int)CC);
    transpose_cvt<<<tg, dim3(32, 8), 0, stream>>>(valid,  xv, (int)NH, (int)CC);

    // 2) pqk = fc1(pse) ; vqk = fc1(val)
    const int R = (int)(BB * CC);   // 8192 rows
    run_fc1(xp, t0, t1, pqk, Wh, bias, R, stream);
    run_fc1(xv, t0, t1, vqk, Wh, bias, R, stream);

    // 3) softmaxes
    softmax_cols<<<dim3((unsigned)(NH / 256), (unsigned)BB), 256, 0, stream>>>(pqk, sQp, (int)CC, (int)NH);
    softmax_rows<<<(unsigned)(BB * CC), 256, 0, stream>>>(pqk, sKp, (int)NH);
    softmax_cols<<<dim3((unsigned)(NH / 256), (unsigned)BB), 256, 0, stream>>>(vqk, sQv, (int)CC, (int)NH);
    softmax_rows<<<(unsigned)(BB * CC), 256, 0, stream>>>(vqk, sKv, (int)NH);

    _Float16* att = pqk;   // [16,512,512] fits in pqk region
    _Float16* bx  = vqk;   // [16,512,1024]

    // 4) pseudo branch: att = softmax_c(pqk) x softmax_n(vqk)^T / 1024 ; bx = att @ pse
    launch_gemm(sQp, CC * NH, sKv, CC * NH, att, CC * CC, nullptr,
                (int)CC, (int)CC, (int)NH, (int)NH, (int)NH,
                1.0f / 1024.0f, 0, (int)BB, stream);
    launch_gemm(att, CC * CC, xpT, NH * CC, bx, CC * NH, nullptr,
                (int)CC, (int)NH, (int)CC, (int)CC, (int)CC,
                1.0f, 0, (int)BB, stream);
    run_fc1(bx, t0, t1, bx, Wh, bias, R, stream);
    finalize_k<<<dim3((unsigned)NH, (unsigned)BB), 256, 0, stream>>>(
        bx, pseudo, out, (int)CC, (int)NH, 1.0f);

    // 5) valid branch
    launch_gemm(sQv, CC * NH, sKp, CC * NH, att, CC * CC, nullptr,
                (int)CC, (int)CC, (int)NH, (int)NH, (int)NH,
                1.0f / 1024.0f, 0, (int)BB, stream);
    launch_gemm(att, CC * CC, xvT, NH * CC, bx, CC * NH, nullptr,
                (int)CC, (int)NH, (int)CC, (int)CC, (int)CC,
                1.0f, 0, (int)BB, stream);
    run_fc1(bx, t0, t1, bx, Wh, bias, R, stream);
    finalize_k<<<dim3((unsigned)NH, (unsigned)BB), 256, 0, stream>>>(
        bx, valid, out + SZX, (int)CC, (int)NH, 1.0f);
}
